// TS3_52682068853399
// MI455X (gfx1250) — compile-verified, matrix operand order checked by reading, not run on hardware
//
#include <hip/hip_runtime.h>

typedef __attribute__((ext_vector_type(16))) __bf16 bf16x16;
typedef __attribute__((ext_vector_type(8)))  float  v8f;

#define NTOK 131072
#define TILE 128

// ---------------- fragment helpers (wave32, 16x16x32 bf16 WMMA) ----------------
// Layout per ISA 7.12.2: lane l holds row/col (l&15), K-chunks [kh,kh+8) and
// [kh+16,kh+24) with kh = (l>>4)*8.

// rows of the lane dimension are contiguous in K (row-major MxK for A, or
// N-major NxK "pre-transposed B").  ld = row stride in elements (mult of 8).
__device__ inline bf16x16 load_frag_rk(const __bf16* base, int ld) {
  int lane = threadIdx.x & 31;
  int r = lane & 15;
  int kh = (lane >> 4) * 8;
  const __bf16* p = base + (size_t)r * ld + kh;
  union { bf16x16 v; uint4 q[2]; } f;
  f.q[0] = *(const uint4*)(p);
  f.q[1] = *(const uint4*)(p + 16);
  return f.v;
}

// K-major storage: element (k, c) at base[k*ld + c]; lane dim strided.
__device__ inline bf16x16 load_frag_col(const __bf16* base, int ld) {
  int lane = threadIdx.x & 31;
  int c = lane & 15;
  int kb = (lane >> 4) * 8;
  union { bf16x16 v; __bf16 h[16]; } f;
#pragma unroll
  for (int j = 0; j < 8; ++j) {
    f.h[j]     = base[(size_t)(kb + j) * ld + c];
    f.h[8 + j] = base[(size_t)(kb + 16 + j) * ld + c];
  }
  return f.v;
}

__device__ inline v8f wmma_bf16(bf16x16 a, bf16x16 b, v8f c) {
  return __builtin_amdgcn_wmma_f32_16x16x32_bf16(false, a, false, b, (short)0, c,
                                                 false, false);
}

__device__ inline float gelu_exact(float v) {
  return 0.5f * v * (1.f + erff(v * 0.70710678118654752f));
}
__device__ inline float silu_f(float v) { return v / (1.f + expf(-v)); }

// ---------------- weight prep: W[K][N] f32 -> WT[Np][Kp] bf16 (zero padded) ----
__global__ __launch_bounds__(256) void k_prep_wt(const float* __restrict__ W,
                                                 __bf16* __restrict__ WT,
                                                 int K, int Nd, int Kp, int Np) {
  int idx = blockIdx.x * 256 + threadIdx.x;
  if (idx >= Kp * Np) return;
  int n = idx / Kp, k = idx % Kp;
  float v = (k < K && n < Nd) ? W[(size_t)k * Nd + n] : 0.f;
  WT[idx] = (__bf16)v;
}

// ---------------- small prep: c, per-layer ada, final ada, zero accumulators --
__global__ __launch_bounds__(256) void k_prep_small(
    const float* t_val, const float* dt_val,
    const float* te_w1, const float* te_b1, const float* te_w2, const float* te_b2,
    const float* de_w1, const float* de_b1, const float* de_w2, const float* de_b2,
    const float* ada_w, const float* ada_b,
    const float* fl_ada_w, const float* fl_ada_b,
    float* adaBuf, float* flAdaBuf, float* tokacc, float* normacc) {
  __shared__ float et[256], ed[256], at[128], ad2[128], cs[128];
  int tid = threadIdx.x;
  if (tid < 128) {
    float fr = expf(-logf(10000.f) * (float)tid / 128.f);
    float t = t_val[0], dt = dt_val[0];
    et[tid] = cosf(t * fr);  et[128 + tid] = sinf(t * fr);
    ed[tid] = cosf(dt * fr); ed[128 + tid] = sinf(dt * fr);
  }
  __syncthreads();
  if (tid < 128) {
    float s1 = te_b1[tid], s2 = de_b1[tid];
    for (int k = 0; k < 256; ++k) {
      s1 += et[k] * te_w1[k * 128 + tid];
      s2 += ed[k] * de_w1[k * 128 + tid];
    }
    at[tid] = silu_f(s1);
    ad2[tid] = silu_f(s2);
  }
  __syncthreads();
  if (tid < 128) {
    float c = te_b2[tid] + de_b2[tid];
    for (int k = 0; k < 128; ++k)
      c += at[k] * te_w2[k * 128 + tid] + ad2[k] * de_w2[k * 128 + tid];
    cs[tid] = silu_f(c);
  }
  __syncthreads();
  for (int idx = tid; idx < 5 * 768; idx += 256) {
    int l = idx / 768, col = idx % 768;
    const float* W = ada_w + (size_t)l * 128 * 768;
    float s = ada_b[idx];
    for (int k = 0; k < 128; ++k) s += cs[k] * W[k * 768 + col];
    adaBuf[idx] = s;
  }
  {
    float s = fl_ada_b[tid];
    for (int k = 0; k < 128; ++k) s += cs[k] * fl_ada_w[k * 256 + tid];
    flAdaBuf[tid] = s;
  }
  for (int idx = tid; idx < 4096; idx += 256) tokacc[idx] = 0.f;
  normacc[tid] = 0.f;
}

// ---------------- embedding: GELU(x@W1+b1)@W2+b2 -> h0 (bf16) ------------------
__global__ __launch_bounds__(256) void k_embed(
    const float* __restrict__ x, const __bf16* __restrict__ w1T,
    const float* __restrict__ b1, const __bf16* __restrict__ w2T,
    const float* __restrict__ b2, __bf16* __restrict__ h0) {
  __shared__ __align__(16) __bf16 xe[TILE][32];
  __shared__ __align__(16) __bf16 g[TILE][264];
  int tid = threadIdx.x, wave = tid >> 5, lane = tid & 31;
  int n0 = blockIdx.x * TILE;
  for (int idx = tid; idx < TILE * 32; idx += 256) {
    int r = idx >> 5, k = idx & 31;
    xe[r][k] = (k < 6) ? (__bf16)x[(size_t)(n0 + r) * 6 + k] : (__bf16)0.f;
  }
  __syncthreads();
  int mt = wave;
  for (int nt = 0; nt < 16; ++nt) {
    v8f acc = {};
    bf16x16 a = load_frag_rk(&xe[mt * 16][0], 32);
    bf16x16 b = load_frag_rk(w1T + (size_t)(nt * 16) * 32, 32);
    acc = wmma_bf16(a, b, acc);
    int col = nt * 16 + (lane & 15);
    int rb = mt * 16 + ((lane >> 4) << 3);
    float bias = b1[col];
#pragma unroll
    for (int r = 0; r < 8; ++r) g[rb + r][col] = (__bf16)gelu_exact(acc[r] + bias);
  }
  __syncthreads();
  for (int nt = 0; nt < 8; ++nt) {
    v8f acc = {};
#pragma unroll
    for (int kk = 0; kk < 8; ++kk) {
      bf16x16 a = load_frag_rk(&g[mt * 16][kk * 32], 264);
      bf16x16 b = load_frag_rk(w2T + (size_t)(nt * 16) * 256 + kk * 32, 256);
      acc = wmma_bf16(a, b, acc);
    }
    int col = nt * 16 + (lane & 15);
    int rb = mt * 16 + ((lane >> 4) << 3);
    float bias = b2[col];
#pragma unroll
    for (int r = 0; r < 8; ++r)
      h0[(size_t)(n0 + rb + r) * 128 + col] = (__bf16)(acc[r] + bias);
  }
}

// ---------------- pass A: LN1+mod, kv GEMM, temp/scores/softmax, pooled reduce -
__global__ __launch_bounds__(256) void k_passA(
    const __bf16* __restrict__ hin, const float* __restrict__ ln1w,
    const float* __restrict__ ln1b, const float* __restrict__ ada,
    const __bf16* __restrict__ kv_wT, const float* __restrict__ kv_b,
    const float* __restrict__ temp_w, const float* __restrict__ temp_b,
    const float* __restrict__ xq, __bf16* __restrict__ wtsOut,
    float* __restrict__ tokacc, float* __restrict__ normacc) {
  __shared__ __align__(16) __bf16 hm[TILE][136];
  __shared__ __align__(16) __bf16 kvs[TILE][264];
  int tid = threadIdx.x, wave = tid >> 5, lane = tid & 31;
  int n0 = blockIdx.x * TILE;

  for (int idx = tid; idx < TILE * 128; idx += 256) {
    int r = idx >> 7, k = idx & 127;
    hm[r][k] = hin[(size_t)(n0 + r) * 128 + k];
  }
  __syncthreads();
  if (tid < TILE) {  // LayerNorm + adaLN modulation (per token)
    float mu = 0.f, m2 = 0.f;
    for (int k = 0; k < 128; ++k) { float v = (float)hm[tid][k]; mu += v; m2 += v * v; }
    mu *= 0.0078125f;
    float rs = rsqrtf(m2 * 0.0078125f - mu * mu + 1e-5f);
    for (int k = 0; k < 128; ++k) {
      float v = ((float)hm[tid][k] - mu) * rs * ln1w[k] + ln1b[k];
      hm[tid][k] = (__bf16)(v * (1.f + ada[128 + k]) + ada[k]);
    }
  }
  __syncthreads();
  {  // kv = hm @ kv_w + b   [128x128]@[128x256]
    int mt = wave;
    for (int nt = 0; nt < 16; ++nt) {
      v8f acc = {};
#pragma unroll
      for (int kk = 0; kk < 4; ++kk) {
        bf16x16 a = load_frag_rk(&hm[mt * 16][kk * 32], 136);
        bf16x16 b = load_frag_rk(kv_wT + (size_t)(nt * 16) * 128 + kk * 32, 128);
        acc = wmma_bf16(a, b, acc);
      }
      int col = nt * 16 + (lane & 15);
      int rb = mt * 16 + ((lane >> 4) << 3);
      float bias = kv_b[col];
#pragma unroll
      for (int r = 0; r < 8; ++r) kvs[rb + r][col] = (__bf16)(acc[r] + bias);
    }
  }
  __syncthreads();
  if (tid < TILE) {  // temperature + scores + softmax over 32 slices, per head
    int r = tid;
    float tp[8];
#pragma unroll
    for (int h = 0; h < 8; ++h) tp[h] = temp_b[h];
    for (int k = 0; k < 128; ++k) {
      float hv = (float)hm[r][k];
#pragma unroll
      for (int h = 0; h < 8; ++h) tp[h] += hv * temp_w[k * 8 + h];
    }
#pragma unroll
    for (int h = 0; h < 8; ++h) {
      float z = tp[h];
      tp[h] = 0.5f + ((z > 20.f) ? z : log1pf(expf(z)));
    }
    for (int h = 0; h < 8; ++h) {
      float xk[16];
#pragma unroll
      for (int d = 0; d < 16; ++d) xk[d] = (float)kvs[r][h * 16 + d];
      float invt = 1.f / tp[h];
      float sc[32], mx = -3.4e38f;
      for (int m = 0; m < 32; ++m) {
        const float* q = xq + (size_t)(h * 32 + m) * 16;
        float s = 0.f;
#pragma unroll
        for (int d = 0; d < 16; ++d) s += q[d] * xk[d];
        s *= invt;
        sc[m] = s;
        mx = fmaxf(mx, s);
      }
      float se = 0.f;
      for (int m = 0; m < 32; ++m) { sc[m] = expf(sc[m] - mx); se += sc[m]; }
      float inv = 1.f / se;
      __bf16* wrow = wtsOut + (size_t)(n0 + r) * 256 + h * 32;
      for (int m = 0; m < 32; ++m) wrow[m] = (__bf16)(sc[m] * inv);
    }
  }
  __syncthreads();
  {  // norm partial: sum over this block's tokens
    int h = tid >> 5, m = tid & 31;
    const __bf16* base = wtsOut + (size_t)n0 * 256 + h * 32 + m;
    float s = 0.f;
    for (int t = 0; t < TILE; ++t) s += (float)base[(size_t)t * 256];
    atomicAdd(&normacc[tid], s);
  }
  // tok partial: wts[m, n_tile] @ xv[n_tile, d]  (2 tiles per wave)
  for (int t = wave; t < 16; t += 8) {
    int h = t & 7, mt = t >> 3;
    v8f acc = {};
#pragma unroll
    for (int kk = 0; kk < 4; ++kk) {
      bf16x16 a =
          load_frag_col(wtsOut + (size_t)(n0 + kk * 32) * 256 + h * 32 + mt * 16, 256);
      bf16x16 b = load_frag_col(&kvs[kk * 32][128 + h * 16], 264);
      acc = wmma_bf16(a, b, acc);
    }
    int d = lane & 15;
    int mb = mt * 16 + ((lane >> 4) << 3);
#pragma unroll
    for (int r = 0; r < 8; ++r)
      atomicAdd(&tokacc[(size_t)(h * 32 + mb + r) * 16 + d], acc[r]);
  }
}

// ---------------- pass B: finalize slice tokens (tiny), zero accs -------------
__global__ __launch_bounds__(256) void k_passB(float* __restrict__ tokacc,
                                               float* __restrict__ normacc,
                                               const float* __restrict__ qkv_w,
                                               __bf16* __restrict__ otT) {
  int tid = threadIdx.x;
  int h = tid >> 5, m = tid & 31;
  float t[16];
  float inv = 1.f / (normacc[tid] + 1e-5f);
#pragma unroll
  for (int d = 0; d < 16; ++d) t[d] = tokacc[tid * 16 + d] * inv;
  float q[16], kk[16], v[16];
  for (int e = 0; e < 16; ++e) {
    float sq = 0.f, sk = 0.f, sv = 0.f;
#pragma unroll
    for (int d = 0; d < 16; ++d) {
      const float* w = qkv_w + (size_t)h * 768 + d * 48;
      sq += t[d] * w[e];
      sk += t[d] * w[16 + e];
      sv += t[d] * w[32 + e];
    }
    q[e] = sq; kk[e] = sk; v[e] = sv;
  }
  float a[16], mx = -3.4e38f;
#pragma unroll
  for (int d = 0; d < 16; ++d) { a[d] = q[d] * kk[d] * 0.25f; mx = fmaxf(mx, a[d]); }
  float se = 0.f;
#pragma unroll
  for (int d = 0; d < 16; ++d) { a[d] = expf(a[d] - mx); se += a[d]; }
  float is = 1.f / se;
#pragma unroll
  for (int d = 0; d < 16; ++d)
    otT[(size_t)h * 512 + d * 32 + m] = (__bf16)(a[d] * is * v[d]);  // [h][d][m]
  __syncthreads();
  for (int idx = tid; idx < 4096; idx += 256) tokacc[idx] = 0.f;
  normacc[tid] = 0.f;
}

// ---------------- pass C: scatter + out proj + residual + MLP -----------------
__global__ __launch_bounds__(256) void k_passC(
    const __bf16* __restrict__ hin, __bf16* __restrict__ hout,
    const float* __restrict__ ln2w, const float* __restrict__ ln2b,
    const float* __restrict__ ada, const __bf16* __restrict__ wts,
    const __bf16* __restrict__ otT, const __bf16* __restrict__ out_wT,
    const float* __restrict__ out_b, const __bf16* __restrict__ m1T,
    const float* __restrict__ m1b, const __bf16* __restrict__ m2T,
    const float* __restrict__ m2b) {
  __shared__ __align__(16) unsigned char smemC[112640];
  typedef __bf16 row136[136];
  row136* xold = (row136*)smemC;             // reused as m1 in phase 4/5
  row136* bufA = (row136*)(smemC + 34816);   // attn out, then hm2
  row136* xmid = (row136*)(smemC + 69632);
  __bf16* otl = (__bf16*)(smemC + 104448);   // 8 KB copy of out_tokT
  row136* m1 = xold;
  int tid = threadIdx.x, wave = tid >> 5, lane = tid & 31;
  int n0 = blockIdx.x * TILE;
  int mt = wave;

  for (int idx = tid; idx < TILE * 128; idx += 256) {
    int r = idx >> 7, k = idx & 127;
    xold[r][k] = hin[(size_t)(n0 + r) * 128 + k];
  }
  for (int idx = tid; idx < 4096; idx += 256) otl[idx] = otT[idx];
  __syncthreads();
  // phase 1: attn[n, h*16+d] = sum_m wts[n,h,m] * out_tok[h,m,d]  (K=32)
  for (int h = 0; h < 8; ++h) {
    v8f acc = {};
    bf16x16 a = load_frag_rk(wts + (size_t)(n0 + mt * 16) * 256 + h * 32, 256);
    bf16x16 b = load_frag_rk(&otl[h * 512], 32);
    acc = wmma_bf16(a, b, acc);
    int col = lane & 15;
    int rb = mt * 16 + ((lane >> 4) << 3);
#pragma unroll
    for (int r = 0; r < 8; ++r) bufA[rb + r][h * 16 + col] = (__bf16)acc[r];
  }
  __syncthreads();
  // phase 2: x = h + g1 * (attn @ out_w + out_b)
  for (int nt = 0; nt < 8; ++nt) {
    v8f acc = {};
#pragma unroll
    for (int kk = 0; kk < 4; ++kk) {
      bf16x16 a = load_frag_rk(&bufA[mt * 16][kk * 32], 136);
      bf16x16 b = load_frag_rk(out_wT + (size_t)(nt * 16) * 128 + kk * 32, 128);
      acc = wmma_bf16(a, b, acc);
    }
    int col = nt * 16 + (lane & 15);
    int rb = mt * 16 + ((lane >> 4) << 3);
    float g1 = ada[256 + col], bias = out_b[col];
#pragma unroll
    for (int r = 0; r < 8; ++r)
      xmid[rb + r][col] = (__bf16)((float)xold[rb + r][col] + g1 * (acc[r] + bias));
  }
  __syncthreads();
  if (tid < TILE) {  // LN2 + modulation
    float mu = 0.f, m2 = 0.f;
    for (int k = 0; k < 128; ++k) { float v = (float)xmid[tid][k]; mu += v; m2 += v * v; }
    mu *= 0.0078125f;
    float rs = rsqrtf(m2 * 0.0078125f - mu * mu + 1e-5f);
    for (int k = 0; k < 128; ++k) {
      float v = ((float)xmid[tid][k] - mu) * rs * ln2w[k] + ln2b[k];
      bufA[tid][k] = (__bf16)(v * (1.f + ada[512 + k]) + ada[384 + k]);
    }
  }
  __syncthreads();
  // phase 4: m1 = gelu(hm2 @ mlp_w1 + b1)
  for (int nt = 0; nt < 8; ++nt) {
    v8f acc = {};
#pragma unroll
    for (int kk = 0; kk < 4; ++kk) {
      bf16x16 a = load_frag_rk(&bufA[mt * 16][kk * 32], 136);
      bf16x16 b = load_frag_rk(m1T + (size_t)(nt * 16) * 128 + kk * 32, 128);
      acc = wmma_bf16(a, b, acc);
    }
    int col = nt * 16 + (lane & 15);
    int rb = mt * 16 + ((lane >> 4) << 3);
    float bias = m1b[col];
#pragma unroll
    for (int r = 0; r < 8; ++r) m1[rb + r][col] = (__bf16)gelu_exact(acc[r] + bias);
  }
  __syncthreads();
  // phase 5: hout = xmid + g2 * (m1 @ mlp_w2 + b2)
  for (int nt = 0; nt < 8; ++nt) {
    v8f acc = {};
#pragma unroll
    for (int kk = 0; kk < 4; ++kk) {
      bf16x16 a = load_frag_rk(&m1[mt * 16][kk * 32], 136);
      bf16x16 b = load_frag_rk(m2T + (size_t)(nt * 16) * 128 + kk * 32, 128);
      acc = wmma_bf16(a, b, acc);
    }
    int col = nt * 16 + (lane & 15);
    int rb = mt * 16 + ((lane >> 4) << 3);
    float g2 = ada[640 + col], bias = m2b[col];
#pragma unroll
    for (int r = 0; r < 8; ++r)
      hout[(size_t)(n0 + rb + r) * 128 + col] =
          (__bf16)((float)xmid[rb + r][col] + g2 * (acc[r] + bias));
  }
}

// ---------------- final layer: LN + mod + [128x6] projection ------------------
__global__ __launch_bounds__(256) void k_final(
    const __bf16* __restrict__ h, const float* __restrict__ lnw,
    const float* __restrict__ lnb, const float* __restrict__ flAda,
    const float* __restrict__ fw, const float* __restrict__ fb,
    float* __restrict__ out) {
  int n = blockIdx.x * 256 + threadIdx.x;
  if (n >= NTOK) return;
  const __bf16* hp = h + (size_t)n * 128;
  float mu = 0.f, m2 = 0.f;
  for (int k = 0; k < 128; ++k) { float v = (float)hp[k]; mu += v; m2 += v * v; }
  mu *= 0.0078125f;
  float rs = rsqrtf(m2 * 0.0078125f - mu * mu + 1e-5f);
  float o[6];
#pragma unroll
  for (int j = 0; j < 6; ++j) o[j] = fb[j];
  for (int k = 0; k < 128; ++k) {
    float v = ((float)hp[k] - mu) * rs * lnw[k] + lnb[k];
    v = v * (1.f + flAda[128 + k]) + flAda[k];
#pragma unroll
    for (int j = 0; j < 6; ++j) o[j] += v * fw[k * 6 + j];
  }
#pragma unroll
  for (int j = 0; j < 6; ++j) out[(size_t)n * 6 + j] = o[j];
}

// ------------------------------- host driver ----------------------------------
extern "C" void kernel_launch(void* const* d_in, const int* in_sizes, int n_in,
                              void* d_out, int out_size, void* d_ws, size_t ws_size,
                              hipStream_t stream) {
  (void)in_sizes; (void)n_in; (void)out_size; (void)ws_size;
  const float* x      = (const float*)d_in[0];
  const float* t_val  = (const float*)d_in[1];
  const float* dt_val = (const float*)d_in[2];
  const float* xe_w1  = (const float*)d_in[3];
  const float* xe_b1  = (const float*)d_in[4];
  const float* xe_w2  = (const float*)d_in[5];
  const float* xe_b2  = (const float*)d_in[6];
  const float* te_w1  = (const float*)d_in[7];
  const float* te_b1  = (const float*)d_in[8];
  const float* te_w2  = (const float*)d_in[9];
  const float* te_b2  = (const float*)d_in[10];
  const float* de_w1  = (const float*)d_in[11];
  const float* de_b1  = (const float*)d_in[12];
  const float* de_w2  = (const float*)d_in[13];
  const float* de_b2  = (const float*)d_in[14];
  const float* ln1_w  = (const float*)d_in[15];
  const float* ln1_b  = (const float*)d_in[16];
  const float* ln2_w  = (const float*)d_in[17];
  const float* ln2_b  = (const float*)d_in[18];
  const float* temp_w = (const float*)d_in[19];
  const float* temp_b = (const float*)d_in[20];
  const float* kv_w   = (const float*)d_in[21];
  const float* kv_b   = (const float*)d_in[22];
  const float* xq     = (const float*)d_in[23];
  const float* qkv    = (const float*)d_in[24];
  const float* out_w  = (const float*)d_in[25];
  const float* out_b  = (const float*)d_in[26];
  const float* ada_w  = (const float*)d_in[27];
  const float* ada_b  = (const float*)d_in[28];
  const float* mlp_w1 = (const float*)d_in[29];
  const float* mlp_b1 = (const float*)d_in[30];
  const float* mlp_w2 = (const float*)d_in[31];
  const float* mlp_b2 = (const float*)d_in[32];
  const float* fl_ln_w  = (const float*)d_in[33];
  const float* fl_ln_b  = (const float*)d_in[34];
  const float* fl_ada_w = (const float*)d_in[35];
  const float* fl_ada_b = (const float*)d_in[36];
  const float* fl_w   = (const float*)d_in[37];
  const float* fl_b   = (const float*)d_in[38];

  char* ws = (char*)d_ws;
  size_t off = 0;
  auto carve = [&](size_t bytes) -> char* {
    char* p = ws + off;
    off += (bytes + 255) & ~(size_t)255;
    return p;
  };
  __bf16* wtsBuf = (__bf16*)carve((size_t)NTOK * 256 * 2);
  __bf16* h0     = (__bf16*)carve((size_t)NTOK * 128 * 2);
  __bf16* h1     = (__bf16*)carve((size_t)NTOK * 128 * 2);
  __bf16* xe1T   = (__bf16*)carve(256 * 32 * 2);
  __bf16* xe2T   = (__bf16*)carve(128 * 256 * 2);
  __bf16* kvT    = (__bf16*)carve((size_t)5 * 256 * 128 * 2);
  __bf16* outT   = (__bf16*)carve((size_t)5 * 128 * 128 * 2);
  __bf16* m1T    = (__bf16*)carve((size_t)5 * 128 * 128 * 2);
  __bf16* m2T    = (__bf16*)carve((size_t)5 * 128 * 128 * 2);
  float* adaB    = (float*)carve(5 * 768 * 4);
  float* fladaB  = (float*)carve(256 * 4);
  float* tokacc  = (float*)carve(4096 * 4);
  float* normacc = (float*)carve(256 * 4);
  __bf16* otT    = (__bf16*)carve(4096 * 2);

  auto prep = [&](const float* W, __bf16* WT, int K, int Nd, int Kp, int Np) {
    int tot = Kp * Np;
    k_prep_wt<<<(tot + 255) / 256, 256, 0, stream>>>(W, WT, K, Nd, Kp, Np);
  };
  prep(xe_w1, xe1T, 6, 256, 32, 256);
  prep(xe_w2, xe2T, 256, 128, 256, 128);
  for (int i = 0; i < 5; ++i) {
    prep(kv_w + (size_t)i * 128 * 256, kvT + (size_t)i * 256 * 128, 128, 256, 128, 256);
    prep(out_w + (size_t)i * 128 * 128, outT + (size_t)i * 128 * 128, 128, 128, 128, 128);
    prep(mlp_w1 + (size_t)i * 128 * 128, m1T + (size_t)i * 128 * 128, 128, 128, 128, 128);
    prep(mlp_w2 + (size_t)i * 128 * 128, m2T + (size_t)i * 128 * 128, 128, 128, 128, 128);
  }
  k_prep_small<<<1, 256, 0, stream>>>(t_val, dt_val, te_w1, te_b1, te_w2, te_b2,
                                      de_w1, de_b1, de_w2, de_b2, ada_w, ada_b,
                                      fl_ada_w, fl_ada_b, adaB, fladaB, tokacc,
                                      normacc);
  k_embed<<<NTOK / TILE, 256, 0, stream>>>(x, xe1T, xe_b1, xe2T, xe_b2, h0);

  for (int i = 0; i < 5; ++i) {
    __bf16* hin  = (i & 1) ? h1 : h0;
    __bf16* hout = (i & 1) ? h0 : h1;
    k_passA<<<NTOK / TILE, 256, 0, stream>>>(
        hin, ln1_w + i * 128, ln1_b + i * 128, adaB + i * 768,
        kvT + (size_t)i * 256 * 128, kv_b + i * 256, temp_w + (size_t)i * 128 * 8,
        temp_b + i * 8, xq + (size_t)i * 8 * 32 * 16, wtsBuf, tokacc, normacc);
    k_passB<<<1, 256, 0, stream>>>(tokacc, normacc, qkv + (size_t)i * 8 * 16 * 48, otT);
    k_passC<<<NTOK / TILE, 256, 0, stream>>>(
        hin, hout, ln2_w + i * 128, ln2_b + i * 128, adaB + i * 768, wtsBuf, otT,
        outT + (size_t)i * 128 * 128, out_b + i * 128, m1T + (size_t)i * 128 * 128,
        mlp_b1 + i * 128, m2T + (size_t)i * 128 * 128, mlp_b2 + i * 128);
  }
  k_final<<<NTOK / 256, 256, 0, stream>>>(h1, fl_ln_w, fl_ln_b, fladaB, fl_w, fl_b,
                                          (float*)d_out);
}
